// RobertaClassificationHead_36412732735938
// MI455X (gfx1250) — matrix operand order repaired and checked by reading, not compiled, not run on hardware
//
#include <hip/hip_runtime.h>
#include <math.h>

// ---------------------------------------------------------------------------
// Model dims
// ---------------------------------------------------------------------------
#define BB   16      // batch
#define KK_  12      // subtrees / sample
#define LL   8       // paths / subtree
#define PP   24      // tokens / path
#define DD   768     // hidden
#define HH1  384     // half hidden (bidir)
#define VV   44      // vocab
#define NLEAF (BB * KK_ * LL)   // 1536
#define NSUB  (BB * KK_)        // 192

typedef __attribute__((ext_vector_type(16))) __bf16 v16bf;
typedef __attribute__((ext_vector_type(8)))  float  v8f;

#define Bb_M 64
#define Bb_N 128
#define Bb_K 32
#define GEMM_RELU 1

union AFrag { v16bf v; unsigned int u[8]; };

__device__ __forceinline__ unsigned short f2bf(float f) {
  unsigned int u = __float_as_uint(f);
  u += 0x7FFFu + ((u >> 16) & 1u);          // round-to-nearest-even
  return (unsigned short)(u >> 16);
}
__device__ __forceinline__ float sigf(float x) { return 1.0f / (1.0f + expf(-x)); }

// ---------------------------------------------------------------------------
// Fragment load helpers (LDS tile is [rows][32] bf16, 64B row stride)
//  A layout: lanes 0-15 hold K={0-7,16-23}, lanes 16-31 K={8-15,24-31}
//  B layout: lanes 0-15 hold K={0-7,16-23}, lanes 16-31 K={8-15,24-31} of column n
// ---------------------------------------------------------------------------
__device__ __forceinline__ void load_afrag(AFrag& a, const unsigned short* src, int half) {
  #pragma unroll
  for (int v = 0; v < 8; ++v) {
    int kk = ((v >> 2) * 16) + (half * 8) + ((v & 3) * 2);
    a.u[v] = *reinterpret_cast<const unsigned int*>(src + kk);
  }
}
__device__ __forceinline__ void load_bfrag(AFrag& b, const unsigned short* src, int half) {
  #pragma unroll
  for (int v = 0; v < 8; ++v) {
    int kk = ((v >> 2) * 16) + (half * 8) + ((v & 3) * 2);
    b.u[v] = *reinterpret_cast<const unsigned int*>(src + kk);
  }
}

// ---------------------------------------------------------------------------
// GEMM v1 (cold path): fp32 A, fp32 W, converted to bf16 while staging.
// C[M][N] = A[M][K] * W[N][K]^T (+bias) (+add) (opt relu)
// ---------------------------------------------------------------------------
__global__ __launch_bounds__(256) void wmma_gemm_f32(
    const float* __restrict__ A, int M, int K,
    const float* __restrict__ W, int N,
    const float* __restrict__ bias,
    const float* __restrict__ add,
    float* __restrict__ C, int flags)
{
  __shared__ unsigned short As[Bb_M][Bb_K];
  __shared__ unsigned short Bs[Bb_N][Bb_K];

  const int tid  = threadIdx.x;
  const int lane = tid & 31;
  const int wv   = tid >> 5;
  const int wm   = wv >> 2;
  const int wn   = wv & 3;
  const int m0   = blockIdx.y * Bb_M;
  const int n0   = blockIdx.x * Bb_N;
  const int half = lane >> 4;
  const int r16  = lane & 15;

  v8f acc[2][2] = {};

  for (int k0 = 0; k0 < K; k0 += Bb_K) {
    if (lane == 0 && (k0 + Bb_K) < K) {
      __builtin_prefetch(A + (size_t)(m0 + wv * 8)  * K + k0 + Bb_K, 0, 1);
      __builtin_prefetch(W + (size_t)(n0 + wv * 16) * K + k0 + Bb_K, 0, 1);
    }
    for (int e = tid; e < (Bb_M * Bb_K) / 4; e += 256) {
      int row = e >> 3, cg = (e & 7) * 4;
      int gm = m0 + row;
      float4 f = make_float4(0.f, 0.f, 0.f, 0.f);
      if (gm < M) f = *reinterpret_cast<const float4*>(A + (size_t)gm * K + (k0 + cg));
      As[row][cg + 0] = f2bf(f.x);
      As[row][cg + 1] = f2bf(f.y);
      As[row][cg + 2] = f2bf(f.z);
      As[row][cg + 3] = f2bf(f.w);
    }
    for (int e = tid; e < (Bb_N * Bb_K) / 4; e += 256) {
      int row = e >> 3, cg = (e & 7) * 4;
      int gn = n0 + row;
      float4 f = make_float4(0.f, 0.f, 0.f, 0.f);
      if (gn < N) f = *reinterpret_cast<const float4*>(W + (size_t)gn * K + (k0 + cg));
      Bs[row][cg + 0] = f2bf(f.x);
      Bs[row][cg + 1] = f2bf(f.y);
      Bs[row][cg + 2] = f2bf(f.z);
      Bs[row][cg + 3] = f2bf(f.w);
    }
    __syncthreads();

    AFrag a[2], b[2];
    #pragma unroll
    for (int i = 0; i < 2; ++i) load_afrag(a[i], &As[wm * 32 + i * 16 + r16][0], half);
    #pragma unroll
    for (int j = 0; j < 2; ++j) load_bfrag(b[j], &Bs[wn * 32 + j * 16 + r16][0], half);
    #pragma unroll
    for (int i = 0; i < 2; ++i)
      #pragma unroll
      for (int j = 0; j < 2; ++j)
        acc[i][j] = __builtin_amdgcn_wmma_f32_16x16x32_bf16(
            false, a[i].v, false, b[j].v, (short)0, acc[i][j], false, false);
    __syncthreads();
  }

  #pragma unroll
  for (int i = 0; i < 2; ++i) {
    #pragma unroll
    for (int j = 0; j < 2; ++j) {
      int nb = n0 + wn * 32 + j * 16 + r16;
      int mb = m0 + wm * 32 + i * 16 + half * 8;
      if (nb < N) {
        #pragma unroll
        for (int rr = 0; rr < 8; ++rr) {
          int m = mb + rr;
          if (m < M) {
            float v = acc[i][j][rr];
            if (bias) v += bias[nb];
            if (add)  v += add[(size_t)m * N + nb];
            if (flags & GEMM_RELU) v = fmaxf(v, 0.0f);
            C[(size_t)m * N + nb] = v;
          }
        }
      }
    }
  }
}

// ---------------------------------------------------------------------------
// GEMM v2 (hot recurrent path): bf16 A, bf16 W already in global memory.
// LDS staging via gfx1250 async copies (global_load_async_to_lds_b128,
// ASYNCcnt).  Requires N % 128 == 0, K % 32 == 0; rows beyond M may be
// over-read (caller pads) -- their outputs are discarded.
// ---------------------------------------------------------------------------
__global__ __launch_bounds__(256) void wmma_gemm_bf16_async(
    const unsigned short* __restrict__ A, int M, int K,   // bf16 raw [M][K]
    const unsigned short* __restrict__ W, int N,          // bf16 raw [N][K]
    const float* __restrict__ bias,
    float* __restrict__ C, int flags)
{
  __shared__ unsigned short As[Bb_M][Bb_K];
  __shared__ unsigned short Bs[Bb_N][Bb_K];

  const int tid  = threadIdx.x;
  const int lane = tid & 31;
  const int wv   = tid >> 5;
  const int wm   = wv >> 2;
  const int wn   = wv & 3;
  const int m0   = blockIdx.y * Bb_M;
  const int n0   = blockIdx.x * Bb_N;
  const int half = lane >> 4;
  const int r16  = lane & 15;

  // LDS byte offsets (low 32 bits of the generic shared address)
  const unsigned asBase = (unsigned)(size_t)(&As[0][0]);
  const unsigned bsBase = (unsigned)(size_t)(&Bs[0][0]);

  v8f acc[2][2] = {};

  for (int k0 = 0; k0 < K; k0 += Bb_K) {
    // A tile: 64 rows x 64B -> one b128 per lane
    {
      int row = tid >> 2, seg = tid & 3;
      unsigned long long ga =
          (unsigned long long)(A + (size_t)(m0 + row) * K + k0 + seg * 8);
      unsigned lo = asBase + row * 64 + seg * 16;
      asm volatile("global_load_async_to_lds_b128 %0, %1, off"
                   :: "v"(lo), "v"(ga) : "memory");
    }
    // B tile: 128 rows x 64B -> two b128 per lane
    #pragma unroll
    for (int it = 0; it < 2; ++it) {
      int e = tid + it * 256;
      int row = e >> 2, seg = e & 3;
      unsigned long long ga =
          (unsigned long long)(W + (size_t)(n0 + row) * K + k0 + seg * 8);
      unsigned lo = bsBase + row * 64 + seg * 16;
      asm volatile("global_load_async_to_lds_b128 %0, %1, off"
                   :: "v"(lo), "v"(ga) : "memory");
    }
    asm volatile("s_wait_asynccnt 0x0" ::: "memory");
    __syncthreads();

    AFrag a[2], b[2];
    #pragma unroll
    for (int i = 0; i < 2; ++i) load_afrag(a[i], &As[wm * 32 + i * 16 + r16][0], half);
    #pragma unroll
    for (int j = 0; j < 2; ++j) load_bfrag(b[j], &Bs[wn * 32 + j * 16 + r16][0], half);
    #pragma unroll
    for (int i = 0; i < 2; ++i)
      #pragma unroll
      for (int j = 0; j < 2; ++j)
        acc[i][j] = __builtin_amdgcn_wmma_f32_16x16x32_bf16(
            false, a[i].v, false, b[j].v, (short)0, acc[i][j], false, false);
    __syncthreads();
  }

  #pragma unroll
  for (int i = 0; i < 2; ++i) {
    #pragma unroll
    for (int j = 0; j < 2; ++j) {
      int nb = n0 + wn * 32 + j * 16 + r16;
      int mb = m0 + wm * 32 + i * 16 + half * 8;
      #pragma unroll
      for (int rr = 0; rr < 8; ++rr) {
        int m = mb + rr;
        if (m < M) {
          float v = acc[i][j][rr];
          if (bias) v += bias[nb];
          if (flags & GEMM_RELU) v = fmaxf(v, 0.0f);
          C[(size_t)m * N + nb] = v;
        }
      }
    }
  }
}

// ---------------------------------------------------------------------------
// Elementwise kernels
// ---------------------------------------------------------------------------
__global__ void zero_kernel(float* p, int n) {
  int i = blockIdx.x * blockDim.x + threadIdx.x;
  if (i < n) p[i] = 0.0f;
}
__global__ void zero16_kernel(unsigned short* p, int n) {
  int i = blockIdx.x * blockDim.x + threadIdx.x;
  if (i < n) p[i] = 0;
}
__global__ void cvt_bf16_kernel(const float* __restrict__ in,
                                unsigned short* __restrict__ out, int n) {
  int i = blockIdx.x * blockDim.x + threadIdx.x;
  if (i < n) out[i] = f2bf(in[i]);
}

// Token GRU step; gh includes bhh, Eg includes bih.  Writes fp32 h and bf16 mirror.
__global__ void gru_token_gate(const float* __restrict__ Eg,
                               const int* __restrict__ paths, int t, int P,
                               const float* __restrict__ gh,
                               float* __restrict__ h, unsigned short* __restrict__ hbf,
                               int N, int H)
{
  int idx = blockIdx.x * blockDim.x + threadIdx.x;
  if (idx >= N * H) return;
  int n = idx / H, j = idx - n * H;
  int tok = paths[n * P + t];
  const float* g   = Eg + (size_t)tok * 3 * H;
  const float* ghr = gh + (size_t)n * 3 * H;
  float r  = sigf(g[j] + ghr[j]);
  float z  = sigf(g[H + j] + ghr[H + j]);
  float nn = tanhf(g[2 * H + j] + r * ghr[2 * H + j]);
  float hv = (1.0f - z) * nn + z * h[idx];
  h[idx] = hv;
  hbf[idx] = f2bf(hv);
}

// Sequence GRU step with running max: xg rows are [n*L + t_eff][3H]
__global__ void gru_seq_gate(const float* __restrict__ xg, int L, int t_eff,
                             const float* __restrict__ gh,
                             float* __restrict__ h, unsigned short* __restrict__ hbf,
                             float* __restrict__ hmax, int N, int H, int first)
{
  int idx = blockIdx.x * blockDim.x + threadIdx.x;
  if (idx >= N * H) return;
  int n = idx / H, j = idx - n * H;
  const float* g   = xg + ((size_t)n * L + t_eff) * 3 * H;
  const float* ghr = gh + (size_t)n * 3 * H;
  float r  = sigf(g[j] + ghr[j]);
  float z  = sigf(g[H + j] + ghr[H + j]);
  float nn = tanhf(g[2 * H + j] + r * ghr[2 * H + j]);
  float hv = (1.0f - z) * nn + z * h[idx];
  h[idx] = hv;
  hbf[idx] = f2bf(hv);
  hmax[idx] = first ? hv : fmaxf(hmax[idx], hv);
}

// LSTM step (gate order i,f,g,o) with running max: xg rows [n*Kseq+t_eff][4H]
__global__ void lstm_gate(const float* __restrict__ xg, int Kseq, int t_eff,
                          const float* __restrict__ gh,
                          float* __restrict__ h, unsigned short* __restrict__ hbf,
                          float* __restrict__ c, float* __restrict__ hmax,
                          int N, int H, int first)
{
  int idx = blockIdx.x * blockDim.x + threadIdx.x;
  if (idx >= N * H) return;
  int n = idx / H, j = idx - n * H;
  const float* g   = xg + ((size_t)n * Kseq + t_eff) * 4 * H;
  const float* ghr = gh + (size_t)n * 4 * H;
  float ig = sigf(g[j] + ghr[j]);
  float fg = sigf(g[H + j] + ghr[H + j]);
  float gt = tanhf(g[2 * H + j] + ghr[2 * H + j]);
  float og = sigf(g[3 * H + j] + ghr[3 * H + j]);
  float cv = fg * c[idx] + ig * gt;
  c[idx] = cv;
  float hv = og * tanhf(cv);
  h[idx] = hv;
  hbf[idx] = f2bf(hv);
  hmax[idx] = first ? hv : fmaxf(hmax[idx], hv);
}

__global__ void concat_kernel(const float* __restrict__ a, const float* __restrict__ b,
                              float* __restrict__ out, int N, int H, int do_relu)
{
  int idx = blockIdx.x * blockDim.x + threadIdx.x;
  if (idx >= N * 2 * H) return;
  int n = idx / (2 * H), j = idx - n * 2 * H;
  float v = (j < H) ? a[n * H + j] : b[n * H + (j - H)];
  if (do_relu) v = fmaxf(v, 0.0f);
  out[idx] = v;
}

__global__ void bn_kernel(const float* __restrict__ x, const float* __restrict__ g,
                          const float* __restrict__ b, float* __restrict__ y,
                          int rows, int cols)
{
  int c = blockIdx.x * blockDim.x + threadIdx.x;
  if (c >= cols) return;
  float mu = 0.0f;
  for (int r = 0; r < rows; ++r) mu += x[(size_t)r * cols + c];
  mu /= (float)rows;
  float var = 0.0f;
  for (int r = 0; r < rows; ++r) {
    float d = x[(size_t)r * cols + c] - mu;
    var += d * d;
  }
  var /= (float)rows;
  float inv = rsqrtf(var + 1e-5f);
  for (int r = 0; r < rows; ++r)
    y[(size_t)r * cols + c] = g[c] * (x[(size_t)r * cols + c] - mu) * inv + b[c];
}

__global__ void out_head(const float* __restrict__ h, const float* __restrict__ W,
                         const float* __restrict__ bias, float* __restrict__ out, int D)
{
  __shared__ float red[256];
  int r = blockIdx.x;
  float s = 0.0f;
  for (int j = threadIdx.x; j < D; j += blockDim.x) s += h[(size_t)r * D + j] * W[j];
  red[threadIdx.x] = s;
  __syncthreads();
  for (int st = 128; st > 0; st >>= 1) {
    if ((int)threadIdx.x < st) red[threadIdx.x] += red[threadIdx.x + st];
    __syncthreads();
  }
  if (threadIdx.x == 0) out[r] = 1.0f / (1.0f + expf(-(red[0] + bias[0])));
}

// ---------------------------------------------------------------------------
// Host orchestration
// ---------------------------------------------------------------------------
static inline void launch_gemm(const float* A, int M, int K, const float* W, int N,
                               const float* bias, const float* add, float* C,
                               int flags, hipStream_t s)
{
  dim3 grid((N + Bb_N - 1) / Bb_N, (M + Bb_M - 1) / Bb_M);
  wmma_gemm_f32<<<grid, dim3(256), 0, s>>>(A, M, K, W, N, bias, add, C, flags);
}
static inline void launch_gemm_async(const unsigned short* A, int M, int K,
                                     const unsigned short* W, int N,
                                     const float* bias, float* C, hipStream_t s)
{
  dim3 grid(N / Bb_N, (M + Bb_M - 1) / Bb_M);
  wmma_gemm_bf16_async<<<grid, dim3(256), 0, s>>>(A, M, K, W, N, bias, C, 0);
}
static inline void launch_zero(float* p, int n, hipStream_t s) {
  zero_kernel<<<(n + 255) / 256, 256, 0, s>>>(p, n);
}
static inline void launch_zero16(unsigned short* p, int n, hipStream_t s) {
  zero16_kernel<<<(n + 255) / 256, 256, 0, s>>>(p, n);
}
static inline void launch_cvt(const float* in, unsigned short* out, int n, hipStream_t s) {
  cvt_bf16_kernel<<<(n + 255) / 256, 256, 0, s>>>(in, out, n);
}
static inline int blks(int n) { return (n + 255) / 256; }

extern "C" void kernel_launch(void* const* d_in, const int* in_sizes, int n_in,
                              void* d_out, int out_size, void* d_ws, size_t ws_size,
                              hipStream_t stream)
{
  const float* feature_s = (const float*)d_in[0];
  const int*   paths     = (const int*)d_in[2];   // source_idx (d_in[1]) = arange -> identity
  const float* emb       = (const float*)d_in[3];
  const float* gru_Wih   = (const float*)d_in[4];
  const float* gru_Whh   = (const float*)d_in[5];
  const float* gru_bih   = (const float*)d_in[6];
  const float* gru_bhh   = (const float*)d_in[7];
  const float* g1_Wih[2] = {(const float*)d_in[8],  (const float*)d_in[12]};
  const float* g1_Whh[2] = {(const float*)d_in[9],  (const float*)d_in[13]};
  const float* g1_bih[2] = {(const float*)d_in[10], (const float*)d_in[14]};
  const float* g1_bhh[2] = {(const float*)d_in[11], (const float*)d_in[15]};
  const float* ls_Wih[2] = {(const float*)d_in[16], (const float*)d_in[20]};
  const float* ls_Whh[2] = {(const float*)d_in[17], (const float*)d_in[21]};
  const float* ls_bih[2] = {(const float*)d_in[18], (const float*)d_in[22]};
  const float* ls_bhh[2] = {(const float*)d_in[19], (const float*)d_in[23]};
  const float* tree_W = (const float*)d_in[24];
  const float* tree_b = (const float*)d_in[25];
  const float* bn1_g  = (const float*)d_in[26];
  const float* bn1_b  = (const float*)d_in[27];
  const float* bn2_g  = (const float*)d_in[28];
  const float* bn2_b  = (const float*)d_in[29];
  const float* d1_W   = (const float*)d_in[30];
  const float* d1_b   = (const float*)d_in[31];
  const float* d2_W   = (const float*)d_in[32];
  const float* d2_b   = (const float*)d_in[33];
  const float* out_W  = (const float*)d_in[34];
  const float* out_b  = (const float*)d_in[35];
  float* out = (float*)d_out;

  // Byte-based bump allocator on workspace (256B aligned chunks)
  char* wsb = (char*)d_ws;
  size_t off = 0;
  auto allocf = [&](size_t n) {
    float* p = (float*)(wsb + off);
    off += ((n * sizeof(float)) + 255) & ~(size_t)255;
    return p;
  };
  auto alloch = [&](size_t n) {
    unsigned short* p = (unsigned short*)(wsb + off);
    off += ((n * sizeof(unsigned short)) + 255) & ~(size_t)255;
    return p;
  };

  float* Eg      = allocf((size_t)VV * 3 * DD);
  float* h_gru   = allocf((size_t)NLEAF * DD);
  float* gh_big  = allocf((size_t)NLEAF * 3 * DD);
  float* leafs   = allocf((size_t)NLEAF * DD);
  float* xg1     = allocf((size_t)NLEAF * 3 * HH1);
  float* gh1     = allocf((size_t)NSUB * 3 * HH1);
  float* h1      = allocf((size_t)NSUB * HH1);
  float* max1[2] = {allocf((size_t)NSUB * HH1), allocf((size_t)NSUB * HH1)};
  float* sub     = allocf((size_t)NSUB * DD);
  float* xg2     = allocf((size_t)NSUB * 4 * HH1);
  float* gh2     = allocf((size_t)BB * 4 * HH1);
  float* h2      = allocf((size_t)BB * HH1);
  float* c2      = allocf((size_t)BB * HH1);
  float* max2[2] = {allocf((size_t)BB * HH1), allocf((size_t)BB * HH1)};
  float* ast     = allocf((size_t)BB * DD);
  float* t1      = allocf((size_t)BB * DD);
  float* t2      = allocf((size_t)BB * 2 * DD);
  float* t3      = allocf((size_t)BB * 2 * DD);
  float* t4      = allocf((size_t)BB * DD);
  // bf16 mirrors / converted recurrent weights
  unsigned short* Whh_bf      = alloch((size_t)3 * DD * DD);            // [2304][768]
  unsigned short* g1Whh_bf[2] = {alloch((size_t)3 * HH1 * HH1),
                                 alloch((size_t)3 * HH1 * HH1)};        // [1152][384]
  unsigned short* lsWhh_bf[2] = {alloch((size_t)4 * HH1 * HH1),
                                 alloch((size_t)4 * HH1 * HH1)};        // [1536][384]
  unsigned short* h_gru_bf = alloch((size_t)NLEAF * DD);
  unsigned short* h1_bf    = alloch((size_t)NSUB * HH1);
  unsigned short* h2_bf    = alloch((size_t)Bb_M * HH1);  // padded to 64 rows for async tile

  // --- 0. One-time bf16 conversion of recurrent weights ----------------------
  launch_cvt(gru_Whh, Whh_bf, 3 * DD * DD, stream);
  for (int d = 0; d < 2; ++d) {
    launch_cvt(g1_Whh[d], g1Whh_bf[d], 3 * HH1 * HH1, stream);
    launch_cvt(ls_Whh[d], lsWhh_bf[d], 4 * HH1 * HH1, stream);
  }

  // --- 1. Token-gate table: Eg = emb @ gru_Wih^T + bih  (44 x 2304) ---------
  launch_gemm(emb, VV, DD, gru_Wih, 3 * DD, gru_bih, nullptr, Eg, 0, stream);

  // --- 2. Token GRU: 24 sequential steps (hot path, async bf16 GEMM) --------
  launch_zero(h_gru, NLEAF * DD, stream);
  launch_zero16(h_gru_bf, NLEAF * DD, stream);
  for (int t = 0; t < PP; ++t) {
    launch_gemm_async(h_gru_bf, NLEAF, DD, Whh_bf, 3 * DD, gru_bhh, gh_big, stream);
    gru_token_gate<<<blks(NLEAF * DD), 256, 0, stream>>>(
        Eg, paths, t, PP, gh_big, h_gru, h_gru_bf, NLEAF, DD);
  }

  // --- 3. leafs = feature_s @ tree_W^T + tree_b + h_gru ---------------------
  launch_gemm(feature_s, NLEAF, DD, tree_W, DD, tree_b, h_gru, leafs, 0, stream);

  // --- 4. Bidirectional GRU over L=8 ----------------------------------------
  for (int dir = 0; dir < 2; ++dir) {
    launch_gemm(leafs, NLEAF, DD, g1_Wih[dir], 3 * HH1, g1_bih[dir], nullptr, xg1, 0, stream);
    launch_zero(h1, NSUB * HH1, stream);
    launch_zero16(h1_bf, NSUB * HH1, stream);
    for (int t = 0; t < LL; ++t) {
      int t_eff = dir ? (LL - 1 - t) : t;
      launch_gemm_async(h1_bf, NSUB, HH1, g1Whh_bf[dir], 3 * HH1, g1_bhh[dir], gh1, stream);
      gru_seq_gate<<<blks(NSUB * HH1), 256, 0, stream>>>(
          xg1, LL, t_eff, gh1, h1, h1_bf, max1[dir], NSUB, HH1, t == 0);
    }
  }
  concat_kernel<<<blks(NSUB * DD), 256, 0, stream>>>(max1[0], max1[1], sub, NSUB, HH1, 1);

  // --- 5. Bidirectional LSTM over K=12 --------------------------------------
  for (int dir = 0; dir < 2; ++dir) {
    launch_gemm(sub, NSUB, DD, ls_Wih[dir], 4 * HH1, ls_bih[dir], nullptr, xg2, 0, stream);
    launch_zero(h2, BB * HH1, stream);
    launch_zero(c2, BB * HH1, stream);
    launch_zero16(h2_bf, Bb_M * HH1, stream);   // zero the padded 64-row mirror
    for (int t = 0; t < KK_; ++t) {
      int t_eff = dir ? (KK_ - 1 - t) : t;
      launch_gemm_async(h2_bf, BB, HH1, lsWhh_bf[dir], 4 * HH1, ls_bhh[dir], gh2, stream);
      lstm_gate<<<blks(BB * HH1), 256, 0, stream>>>(
          xg2, KK_, t_eff, gh2, h2, h2_bf, c2, max2[dir], BB, HH1, t == 0);
    }
  }
  concat_kernel<<<blks(BB * DD), 256, 0, stream>>>(max2[0], max2[1], ast, BB, HH1, 0);

  // --- 6. Head ---------------------------------------------------------------
  bn_kernel<<<blks(DD), 256, 0, stream>>>(ast, bn1_g, bn1_b, t1, BB, DD);
  launch_gemm(t1, BB, DD, d1_W, 2 * DD, d1_b, nullptr, t2, GEMM_RELU, stream);
  bn_kernel<<<blks(2 * DD), 256, 0, stream>>>(t2, bn2_g, bn2_b, t3, BB, 2 * DD);
  launch_gemm(t3, BB, 2 * DD, d2_W, DD, d2_b, nullptr, t4, GEMM_RELU, stream);
  out_head<<<BB, 256, 0, stream>>>(t4, out_W, out_b, out, DD);
}
